// RTVLoss_77025943487044
// MI455X (gfx1250) — compile-verified, matrix operand order checked by reading, not run on hardware
//
#include <hip/hip_runtime.h>

typedef __attribute__((ext_vector_type(2)))  float    v2f;
typedef __attribute__((ext_vector_type(8)))  float    v8f;
typedef __attribute__((ext_vector_type(16))) _Float16 v16h;
typedef __attribute__((ext_vector_type(4)))  int      v4i;

namespace {
constexpr float  GKc     = 50.132565492620005f; // 5*sqrt(2*pi)*sigma^2, sigma=2
constexpr float  EPSc    = 1e-6f;
constexpr int    HW      = 1024;
constexpr int    OUTD    = 1020;                // valid 5x5 conv output dim
constexpr int    LROWS   = 20;                  // 16 out rows + 4 halo
constexpr int    LCOLS   = 132;                 // 128 out cols + 4 halo
constexpr int    LSTRIDE = 144;                 // 2*144 % 64 == 32 -> bank-split halves
constexpr int    NBX = 8, NBY = 64, NBZ = 48;   // 48 = B*C images
constexpr int    NPART = NBX * NBY * NBZ;       // 24576 block partials
constexpr double NMEAN = 16.0 * 3.0 * 1020.0 * 1020.0;
}

#if __has_builtin(__builtin_amdgcn_wmma_f32_16x16x4_f32)
  #define RTV_WMMA_F32X4 1
#endif
#if !defined(RTV_WMMA_F32X4) && __has_builtin(__builtin_amdgcn_wmma_f32_16x16x32_f16)
  #define RTV_WMMA_F16 1
#endif
#if __has_builtin(__builtin_amdgcn_global_load_async_to_lds_b128)
  #define RTV_ASYNC 1
#endif

__device__ __forceinline__ void rtv_wait_async0() {
#if __has_builtin(__builtin_amdgcn_s_wait_asynccnt)
  __builtin_amdgcn_s_wait_asynccnt(0);
#else
  asm volatile("s_wait_asynccnt 0" ::: "memory");
#endif
}

// horizontal 5-tap sum (value and abs) from an LDS tile row
__device__ __forceinline__ void hsum5(const float* lds, int r, int lc,
                                      float& s, float& sa) {
  const float* p = lds + r * LSTRIDE + lc;
  float x0 = p[0], x1 = p[1], x2 = p[2], x3 = p[3], x4 = p[4];
  s  = ((x0 + x1) + (x2 + x3)) + x4;
  sa = ((__builtin_fabsf(x0) + __builtin_fabsf(x1)) +
        (__builtin_fabsf(x2) + __builtin_fabsf(x3))) + __builtin_fabsf(x4);
}

__global__ __launch_bounds__(256)
void RTVLoss_main(const float* __restrict__ gh, const float* __restrict__ gw,
                  const float* __restrict__ gam, float* __restrict__ partials) {
  __shared__ float lds[2 * LROWS * LSTRIDE];
  __shared__ float waveSums[8];
  float* ldsH = lds;
  float* ldsW = lds + LROWS * LSTRIDE;

  const int tid   = threadIdx.x;
  const int img   = blockIdx.z;          // b*3 + c
  const int row0  = blockIdx.y * 16;
  const int cbase = blockIdx.x * 128;
  const size_t ibase = (size_t)img * (HW * HW);

  // ---- cooperative halo-tile staging into LDS ----
#if defined(RTV_ASYNC)
  {
    // Async CDNA5 path: 16B chunks direct to LDS, ASYNCcnt-tracked.
    // Each 132-float row = 33 b128 chunks; chunk start cbase+4k is 16B aligned.
    constexpr int CPR    = LCOLS / 4;        // 33 chunks per row
    constexpr int CHUNKS = LROWS * CPR;      // 660 per matrix
    for (int i = tid; i < 2 * CHUNKS; i += 256) {
      const int mat = (i >= CHUNKS) ? 1 : 0;
      const int q   = i - mat * CHUNKS;
      const int r   = q / CPR;
      const int k   = q - r * CPR;
      int grow  = row0 + r;      if (grow > HW - 1) grow = HW - 1;
      int gcolf = cbase + 4 * k; if (gcolf > OUTD)  gcolf = OUTD;   // stay in row; dup only feeds masked cols
      const float* gp = (mat ? gw : gh) + ibase + (size_t)grow * HW + gcolf;
      float*       lp = (mat ? ldsW : ldsH) + r * LSTRIDE + 4 * k;
      __builtin_amdgcn_global_load_async_to_lds_b128(
          (__attribute__((address_space(1))) v4i*)gp,
          (__attribute__((address_space(3))) v4i*)lp,
          /*imm offset*/ 0, /*cpol*/ 0);
    }
    rtv_wait_async0();
  }
#else
  for (int i = tid; i < LROWS * LCOLS; i += 256) {
    int r = i / LCOLS;
    int c = i - r * LCOLS;
    int grow = row0 + r;  if (grow > HW - 1) grow = HW - 1;
    int gcol = cbase + c; if (gcol > HW - 1) gcol = HW - 1;
    size_t off = ibase + (size_t)grow * HW + gcol;
    ldsH[r * LSTRIDE + c] = gh[off];
    ldsW[r * LSTRIDE + c] = gw[off];
  }
#endif
  __syncthreads();

  // ---- per-wave 16x16 output tile ----
  const int lane = tid & 31;
  const int wv   = tid >> 5;
  const int n    = lane & 15;      // output column within tile (B/D layout)
  const int hi   = lane >> 4;      // lane half
  const int lc   = wv * 16 + n;    // column inside LDS tile
  const int m    = n;              // A-matrix row owned by this lane
  const int col0 = cbase + wv * 16;

  v8f accVH = {0,0,0,0,0,0,0,0};
  v8f accAH = accVH, accVW = accVH, accAW = accVH;

#if defined(RTV_WMMA_F32X4)
  // vertical 5-window sum as D = Band(16x20) x Hsum(20x16), K=4 chunks, f32 WMMA
  const int koff = hi * 2;
  #pragma unroll
  for (int j = 0; j < 5; ++j) {
    const int r0 = 4 * j + koff;
    const int r1 = r0 + 1;
    v2f a;
    a.x = (r0 >= m && r0 <= m + 4) ? 1.0f : 0.0f;   // band matrix (exact)
    a.y = (r1 >= m && r1 <= m + 4) ? 1.0f : 0.0f;
    float s0, sa0, s1, sa1;
    hsum5(ldsH, r0, lc, s0, sa0);
    hsum5(ldsH, r1, lc, s1, sa1);
    v2f bv; bv.x = s0;  bv.y = s1;
    v2f ba; ba.x = sa0; ba.y = sa1;
    accVH = __builtin_amdgcn_wmma_f32_16x16x4_f32(false, a, false, bv, (short)0, accVH, false, false);
    accAH = __builtin_amdgcn_wmma_f32_16x16x4_f32(false, a, false, ba, (short)0, accAH, false, false);
    hsum5(ldsW, r0, lc, s0, sa0);
    hsum5(ldsW, r1, lc, s1, sa1);
    bv.x = s0; bv.y = s1; ba.x = sa0; ba.y = sa1;
    accVW = __builtin_amdgcn_wmma_f32_16x16x4_f32(false, a, false, bv, (short)0, accVW, false, false);
    accAW = __builtin_amdgcn_wmma_f32_16x16x4_f32(false, a, false, ba, (short)0, accAW, false, false);
  }
#elif defined(RTV_WMMA_F16)
  // single 16x16x32 f16 WMMA per chain; band exact in f16, rows 20..31 zero
  v16h aM, bVH, bAH, bVW, bAW;
  #pragma unroll
  for (int e = 0; e < 16; ++e) {
    const int K = e + 8 * hi + ((e >= 8) ? 8 : 0);   // lane's K slots
    aM[e] = (_Float16)((K >= m && K <= m + 4) ? 1.0f : 0.0f);
    float sv = 0.f, sa = 0.f, tv = 0.f, ta = 0.f;
    if (K < LROWS) { hsum5(ldsH, K, lc, sv, sa); hsum5(ldsW, K, lc, tv, ta); }
    bVH[e] = (_Float16)sv; bAH[e] = (_Float16)sa;
    bVW[e] = (_Float16)tv; bAW[e] = (_Float16)ta;
  }
  accVH = __builtin_amdgcn_wmma_f32_16x16x32_f16(false, aM, false, bVH, (short)0, accVH, false, false);
  accAH = __builtin_amdgcn_wmma_f32_16x16x32_f16(false, aM, false, bAH, (short)0, accAH, false, false);
  accVW = __builtin_amdgcn_wmma_f32_16x16x32_f16(false, aM, false, bVW, (short)0, accVW, false, false);
  accAW = __builtin_amdgcn_wmma_f32_16x16x32_f16(false, aM, false, bAW, (short)0, accAW, false, false);
#else
  // VALU fallback, fills the same D-layout
  #pragma unroll
  for (int v = 0; v < 8; ++v) {
    const int mr = v + 8 * hi;
    float sv = 0.f, sa = 0.f, tv = 0.f, ta = 0.f, h, ha;
    #pragma unroll
    for (int r = 0; r < 5; ++r) {
      hsum5(ldsH, mr + r, lc, h, ha); sv += h; sa += ha;
      hsum5(ldsW, mr + r, lc, h, ha); tv += h; ta += ha;
    }
    accVH[v] = sv; accAH[v] = sa; accVW[v] = tv; accAW[v] = ta;
  }
#endif

  // ---- per-element loss terms with valid-region masking ----
  const bool colOK = (col0 + n) < OUTD;
  float wsum = 0.f;
  #pragma unroll
  for (int v = 0; v < 8; ++v) {
    const int mr = v + 8 * hi;                       // D layout: lanes>=16 hold M=v+8
    const bool ok = colOK && ((row0 + mr) < OUTD);
    float lx = __builtin_fabsf(GKc * accVH[v]);
    float dx = GKc * accAH[v];
    float ly = __builtin_fabsf(GKc * accVW[v]);
    float dy = GKc * accAW[v];
    float t  = dx / (lx + EPSc) + dy / (ly + EPSc);
    wsum += ok ? t : 0.f;
  }

  // wave32 shuffle reduction, then fixed-order block combine (deterministic)
  #pragma unroll
  for (int off = 16; off > 0; off >>= 1) wsum += __shfl_xor(wsum, off, 32);
  if (lane == 0) waveSums[wv] = wsum;
  __syncthreads();
  if (tid == 0) {
    const int b = img / 3;
    const float wgt = expf(1.0f - gam[b]);
    float s = 0.f;
    #pragma unroll
    for (int i = 0; i < 8; ++i) s += waveSums[i];
    const int bid = (blockIdx.z * NBY + blockIdx.y) * NBX + blockIdx.x;
    partials[bid] = s * wgt;
  }
}

__global__ __launch_bounds__(1024)
void RTVLoss_reduce(const float* __restrict__ partials, float* __restrict__ out) {
  __shared__ double sd[1024];
  double acc = 0.0;
  for (int i = threadIdx.x; i < NPART; i += 1024) acc += (double)partials[i];
  sd[threadIdx.x] = acc;
  __syncthreads();
  for (int st = 512; st > 0; st >>= 1) {
    if (threadIdx.x < st) sd[threadIdx.x] += sd[threadIdx.x + st];
    __syncthreads();
  }
  if (threadIdx.x == 0) out[0] = (float)(sd[0] / NMEAN);
}

extern "C" void kernel_launch(void* const* d_in, const int* in_sizes, int n_in,
                              void* d_out, int out_size, void* d_ws, size_t ws_size,
                              hipStream_t stream) {
  (void)in_sizes; (void)n_in; (void)out_size; (void)ws_size;
  const float* gh  = (const float*)d_in[0];
  const float* gw  = (const float*)d_in[1];
  const float* gam = (const float*)d_in[2];
  float* partials  = (float*)d_ws;            // NPART floats (96 KB)

  dim3 grid(NBX, NBY, NBZ), block(256);
  RTVLoss_main<<<grid, block, 0, stream>>>(gh, gw, gam, partials);
  RTVLoss_reduce<<<1, 1024, 0, stream>>>(partials, (float*)d_out);
}
